// MultiHeadAttention_56118042690041
// MI455X (gfx1250) — matrix operand order repaired and checked by reading, not compiled, not run on hardware
//
#include <hip/hip_runtime.h>
#include <hip/hip_bf16.h>

// MI455X / gfx1250: wave32, WMMA 16x16x32 bf16 -> f32 accumulate, TDM staging.
#define B_  2
#define S_  2048
#define D_  1024
#define H_  16
#define DK_ 64

typedef __attribute__((ext_vector_type(16))) __bf16 v16bf;
typedef __attribute__((ext_vector_type(8)))  float  v8f;

union FragU { v16bf v; unsigned int u[8]; };

#if defined(__HIP_DEVICE_COMPILE__) && defined(__gfx1250__) && \
    __has_builtin(__builtin_amdgcn_tensor_load_to_lds) &&      \
    __has_builtin(__builtin_amdgcn_s_wait_tensorcnt)
#define USE_TDM 1
#else
#define USE_TDM 0
#endif

__device__ __forceinline__ v8f wmma_bf16(v16bf a, v16bf b, v8f c) {
  // D = A(16x32 bf16) * B(32x16 bf16) + C(16x16 f32)
  return __builtin_amdgcn_wmma_f32_16x16x32_bf16(
      /*neg_a=*/false, a, /*neg_b=*/false, b,
      /*c_mod=*/(short)0, c, /*reuse_a=*/false, /*reuse_b=*/false);
}

__device__ __forceinline__ unsigned pack_bf16x2(float a, float b) {
  union { unsigned u; __bf16 h[2]; } p;
  p.h[0] = (__bf16)a; p.h[1] = (__bf16)b;
  return p.u;
}

// A-matrix 16x32 bf16 fragment (ISA 7.12.2): lane holds row M=lane%16.
// VGPR j: j<4 -> K=2j,2j+1 ; j>=4 -> K=2j+8,2j+9 ; +8 for hi half-wave.
__device__ __forceinline__ v16bf load_a_frag(const __bf16* p, int row, int ld,
                                             int kbase, int hi) {
  FragU f;
#pragma unroll
  for (int j = 0; j < 8; ++j) {
    int kk = kbase + 2 * j + (j >= 4 ? 8 : 0) + hi * 8;
    f.u[j] = *(const unsigned int*)(p + row * ld + kk);
  }
  return f.v;
}

// B-matrix 32x16 bf16 fragment: lane holds column N=lane%16.
// VGPR j: j<4 -> K=2j,2j+1 ; j>=4 -> K=2j+8,2j+9 ; +8 for hi half-wave. LDS layout [N][K].
__device__ __forceinline__ v16bf load_b_frag(const __bf16* p, int nrow, int ld,
                                             int kbase, int hi) {
  FragU f;
#pragma unroll
  for (int j = 0; j < 8; ++j) {
    int kk = kbase + 2 * j + (j >= 4 ? 8 : 0) + hi * 8;
    f.u[j] = *(const unsigned int*)(p + nrow * ld + kk);
  }
  return f.v;
}

#if USE_TDM
typedef __attribute__((ext_vector_type(4))) unsigned int v4u;
typedef __attribute__((ext_vector_type(8))) int v8i;
typedef __attribute__((ext_vector_type(4))) int v4i;

// Issue a TDM 2D bf16 tile load Global->LDS with LDS padding (D# per ISA 8.x).
// tile_w/tile_h in elements; row_stride in elements; pad_interval encoded per
// ISA (3 => every 16 dwords, 4 => every 32 dwords); pad_amount_m1 dwords-1.
__device__ __forceinline__ void tdm_load_2d_bf16(const __bf16* gsrc,
                                                 __bf16* ldst,
                                                 unsigned tile_w, unsigned tile_h,
                                                 unsigned row_stride,
                                                 unsigned pad_interval,
                                                 unsigned pad_amount_m1) {
  unsigned long long ga = (unsigned long long)gsrc;
  unsigned lds = (unsigned)(unsigned long long)(void*)ldst;  // addr[31:0] = LDS offset
  v4u g0;
  g0[0] = 1u;                                   // count=1, user descriptor
  g0[1] = lds;                                  // lds_addr
  g0[2] = (unsigned)ga;                         // global_addr[31:0]
  g0[3] = (unsigned)((ga >> 32) & 0x1FFFFFFu)   // global_addr[56:32]
          | (2u << 30);                         // type=2 (image)
  v8i g1;
  unsigned dw0 = (1u << 16)                     // data_size=1 -> 2 bytes
               | (1u << 20)                     // pad_enable
               | (pad_interval << 22)
               | (pad_amount_m1 << 25);
  g1[0] = (int)dw0;
  g1[1] = (int)(tile_w << 16);                  // tensor_dim0[15:0] at bits[63:48]
  g1[2] = (int)(tile_h << 16);                  // tensor_dim1[15:0] at bits[95:80] (dim0 hi=0)
  g1[3] = (int)(tile_w << 16);                  // tile_dim0 at bits[127:112]
  g1[4] = (int)tile_h;                          // tile_dim1 (tile_dim2=0)
  g1[5] = (int)row_stride;                      // tensor_dim0_stride[31:0]
  g1[6] = 0;
  g1[7] = 0;
  v4i g2 = {0, 0, 0, 0};
  v4i g3 = {0, 0, 0, 0};
#if __clang_major__ >= 23
  v8i gx = {0, 0, 0, 0, 0, 0, 0, 0};
  __builtin_amdgcn_tensor_load_to_lds(g0, g1, g2, g3, gx, 0);
#else
  __builtin_amdgcn_tensor_load_to_lds(g0, g1, g2, g3, 0);
#endif
}
#endif  // USE_TDM

// ---------------------------------------------------------------------------
// Kernel 1: per-head projection GEMM  out(BHS,DK) = X(BS,D) @ W(HDK,D)^T + b
// f32 inputs staged to LDS as bf16 via b128 loads + packed dword stores.
// Block tile 128(M) x 64(N), K step 32. 256 threads = 8 waves, wave owns 16x64.
// ---------------------------------------------------------------------------
__global__ __launch_bounds__(256)
void proj_kernel(const float* __restrict__ X, const float* __restrict__ W,
                 const float* __restrict__ bias, __bf16* __restrict__ out) {
  constexpr int AS = 34;                       // padded LDS stride
  __shared__ __bf16 at[128 * AS];
  __shared__ __bf16 wt[64 * AS];
  const int tid  = threadIdx.x;
  const int lane = tid & 31;
  const int wave = tid >> 5;
  const int hi   = (lane >> 4) & 1;
  const int mBase = blockIdx.y * 128;
  const int nBase = blockIdx.x * 64;

  v8f acc[4] = {};

  for (int kb = 0; kb < D_; kb += 32) {
    __syncthreads();
#pragma unroll
    for (int i = 0; i < 4; ++i) {              // A tile: 1024 float4 loads
      int idx = i * 256 + tid;
      int r = idx >> 3, c = (idx & 7) * 4;
      float4 f = *(const float4*)&X[(mBase + r) * D_ + kb + c];
      *(unsigned*)&at[r * AS + c]     = pack_bf16x2(f.x, f.y);
      *(unsigned*)&at[r * AS + c + 2] = pack_bf16x2(f.z, f.w);
    }
#pragma unroll
    for (int i = 0; i < 2; ++i) {              // W tile: 512 float4 loads
      int idx = i * 256 + tid;
      int r = idx >> 3, c = (idx & 7) * 4;
      float4 f = *(const float4*)&W[(nBase + r) * D_ + kb + c];
      *(unsigned*)&wt[r * AS + c]     = pack_bf16x2(f.x, f.y);
      *(unsigned*)&wt[r * AS + c + 2] = pack_bf16x2(f.z, f.w);
    }
    if (kb + 32 < D_) {                        // prefetch next K tile lines
      if (tid < 128)
        __builtin_prefetch(&X[(mBase + tid) * D_ + kb + 32], 0, 3);
      else if (tid < 192)
        __builtin_prefetch(&W[(nBase + tid - 128) * D_ + kb + 32], 0, 3);
    }
    __syncthreads();

    const int mrow = wave * 16 + (lane & 15);
    v16bf a = load_a_frag(at, mrow, AS, 0, hi);
    v16bf bf4[4];
#pragma unroll
    for (int t = 0; t < 4; ++t)
      bf4[t] = load_b_frag(wt, t * 16 + (lane & 15), AS, 0, hi);
#pragma unroll
    for (int t = 0; t < 4; ++t)                // back-to-back WMMA chain
      acc[t] = wmma_bf16(a, bf4[t], acc[t]);
  }

  const int nloc = lane & 15;
#pragma unroll
  for (int t = 0; t < 4; ++t) {
    int n = nBase + t * 16 + nloc;
    float bv = bias[n];
    int h = n >> 6, dk = n & 63;
#pragma unroll
    for (int r = 0; r < 8; ++r) {
      int m = mBase + wave * 16 + r + hi * 8;
      int bb = m >> 11, s = m & (S_ - 1);
      out[(((bb * H_ + h) * S_) + s) * DK_ + dk] = (__bf16)(acc[t][r] + bv);
    }
  }
}

// ---------------------------------------------------------------------------
// Kernel 2: attention per (b,h), faithful to reference bug:
//   scores = v2 @ k2^T / 8 ; attn = softmax ; out = attn @ q2
// Flash-style over 64-key tiles; k2/v2 tiles staged by the Tensor Data Mover
// (pad_interval=32 dwords, pad=1 dword -> stride 66), online softmax with
// 16-lane shfl_xor reductions. 128 threads = 4 waves, wave owns 16 rows.
// ---------------------------------------------------------------------------
__global__ __launch_bounds__(128)
void attn_kernel(const __bf16* __restrict__ q2, const __bf16* __restrict__ k2,
                 const __bf16* __restrict__ v2, __bf16* __restrict__ att) {
  constexpr int LD = 66;                       // 33 dwords = 32 + 1 pad
  __shared__ __bf16 vt[64 * LD];               // v2 rows tile   [row][dk]
  __shared__ __bf16 kt[64 * LD];               // k2 keys tile   [key][dk] == B [N][K]
  __shared__ __bf16 qT[64 * LD];               // q2 transposed  [dk][key] == B [N][K]
  __shared__ __bf16 pt[4 * 16 * LD];           // per-wave P tile 16x64

  const int tid  = threadIdx.x;
  const int lane = tid & 31;
  const int wave = tid >> 5;
  const int hi   = (lane >> 4) & 1;
  const int rowBase = blockIdx.x * 64;
  const int h = blockIdx.y, b = blockIdx.z;
  const __bf16* qh = q2 + ((b * H_ + h) * S_) * DK_;
  const __bf16* kh = k2 + ((b * H_ + h) * S_) * DK_;
  const __bf16* vh = v2 + ((b * H_ + h) * S_) * DK_;

#if USE_TDM
  if (wave == 0) {                             // DMA v2 tile into LDS
    tdm_load_2d_bf16(vh + rowBase * DK_, vt, 64, 64, DK_, /*interval 32dw*/4, 0);
    __builtin_amdgcn_s_wait_tensorcnt(0);
  }
#else
#pragma unroll
  for (int i = 0; i < 8; ++i) {
    int idx = i * 128 + tid;
    int r = idx >> 4, c = (idx & 15) * 4;
    unsigned long long d = *(const unsigned long long*)&vh[(rowBase + r) * DK_ + c];
    *(unsigned*)&vt[r * LD + c]     = (unsigned)d;
    *(unsigned*)&vt[r * LD + c + 2] = (unsigned)(d >> 32);
  }
#endif
  __syncthreads();

  const int mrow = wave * 16 + (lane & 15);
  v16bf a0 = load_a_frag(vt, mrow, LD, 0, hi);
  v16bf a1 = load_a_frag(vt, mrow, LD, 32, hi);

  float rowm[8], rowl[8];
#pragma unroll
  for (int r = 0; r < 8; ++r) { rowm[r] = -3.0e38f; rowl[r] = 0.0f; }
  v8f o[4] = {};
  __bf16* pw = pt + wave * 16 * LD;

  for (int jb = 0; jb < S_ / 64; ++jb) {
    const int keyBase = jb * 64;
    __syncthreads();
#if USE_TDM
    if (wave == 0)                             // DMA k2 tile while ALUs transpose q2
      tdm_load_2d_bf16(kh + keyBase * DK_, kt, 64, 64, DK_, 4, 0);
#endif
#pragma unroll
    for (int i = 0; i < 8; ++i) {              // transpose q2 tile (values matrix)
      int idx = i * 128 + tid;
      int r = idx >> 4, c = (idx & 15) * 4;
      unsigned long long d = *(const unsigned long long*)&qh[(keyBase + r) * DK_ + c];
      qT[(c + 0) * LD + r] = ((const __bf16*)&d)[0];
      qT[(c + 1) * LD + r] = ((const __bf16*)&d)[1];
      qT[(c + 2) * LD + r] = ((const __bf16*)&d)[2];
      qT[(c + 3) * LD + r] = ((const __bf16*)&d)[3];
    }
#if USE_TDM
    if (wave == 0) __builtin_amdgcn_s_wait_tensorcnt(0);
#else
#pragma unroll
    for (int i = 0; i < 8; ++i) {
      int idx = i * 128 + tid;
      int r = idx >> 4, c = (idx & 15) * 4;
      unsigned long long d = *(const unsigned long long*)&kh[(keyBase + r) * DK_ + c];
      *(unsigned*)&kt[r * LD + c]     = (unsigned)d;
      *(unsigned*)&kt[r * LD + c + 2] = (unsigned)(d >> 32);
    }
#endif
    __syncthreads();

    // scores tile: 16 rows x 64 keys, scaled by 1/sqrt(64)
    v16bf kb0[4], kb1[4];
#pragma unroll
    for (int t = 0; t < 4; ++t) {
      kb0[t] = load_b_frag(kt, t * 16 + (lane & 15), LD, 0, hi);
      kb1[t] = load_b_frag(kt, t * 16 + (lane & 15), LD, 32, hi);
    }
    v8f sc[4];
#pragma unroll
    for (int t = 0; t < 4; ++t) {
      v8f z = {};
      z = wmma_bf16(a0, kb0[t], z);
      z = wmma_bf16(a1, kb1[t], z);
      sc[t] = z * 0.125f;
    }

    // online softmax
    float alpha[8], rs[8];
#pragma unroll
    for (int r = 0; r < 8; ++r) {
      float m0 = fmaxf(fmaxf(sc[0][r], sc[1][r]), fmaxf(sc[2][r], sc[3][r]));
#pragma unroll
      for (int off = 8; off >= 1; off >>= 1)
        m0 = fmaxf(m0, __shfl_xor(m0, off, 32));
      float mnew = fmaxf(rowm[r], m0);
      alpha[r] = __expf(rowm[r] - mnew);
      rowm[r] = mnew;
      rs[r] = 0.0f;
    }
#pragma unroll
    for (int t = 0; t < 4; ++t) {
#pragma unroll
      for (int r = 0; r < 8; ++r) {
        float p = __expf(sc[t][r] - rowm[r]);
        rs[r] += p;
        pw[(r + hi * 8) * LD + t * 16 + (lane & 15)] = (__bf16)p;
      }
    }
#pragma unroll
    for (int r = 0; r < 8; ++r) {
#pragma unroll
      for (int off = 8; off >= 1; off >>= 1)
        rs[r] += __shfl_xor(rs[r], off, 32);
      rowl[r] = rowl[r] * alpha[r] + rs[r];
    }
#pragma unroll
    for (int t = 0; t < 4; ++t)
#pragma unroll
      for (int r = 0; r < 8; ++r)
        o[t][r] *= alpha[r];

    // O += P(16x64) @ q2tile(64x64)
    v16bf pa0 = load_a_frag(pw, lane & 15, LD, 0, hi);
    v16bf pa1 = load_a_frag(pw, lane & 15, LD, 32, hi);
    v16bf qb0[4], qb1[4];
#pragma unroll
    for (int t = 0; t < 4; ++t) {
      qb0[t] = load_b_frag(qT, t * 16 + (lane & 15), LD, 0, hi);
      qb1[t] = load_b_frag(qT, t * 16 + (lane & 15), LD, 32, hi);
    }
#pragma unroll
    for (int t = 0; t < 4; ++t) {
      o[t] = wmma_bf16(pa0, qb0[t], o[t]);
      o[t] = wmma_bf16(pa1, qb1[t], o[t]);
    }
  }

  // normalize and write concat-head layout (B,S,H*DK) bf16
#pragma unroll
  for (int t = 0; t < 4; ++t) {
    int dk = t * 16 + (lane & 15);
#pragma unroll
    for (int r = 0; r < 8; ++r) {
      int s = rowBase + wave * 16 + r + hi * 8;
      float val = o[t][r] * (1.0f / rowl[r]);
      att[(b * S_ + s) * D_ + h * DK_ + dk] = (__bf16)val;
    }
  }
}

// ---------------------------------------------------------------------------
// Kernel 3: output projection  out(BS,D) f32 = att(BS,D) bf16 @ Wo^T + bo
// A tile is bf16 in global -> staged by TDM (pad_interval=16 dwords, pad 1).
// ---------------------------------------------------------------------------
__global__ __launch_bounds__(256)
void oproj_kernel(const __bf16* __restrict__ Xb, const float* __restrict__ W,
                  const float* __restrict__ bias, float* __restrict__ out) {
  constexpr int AS = 34;                       // 17 dwords = 16 + 1 pad
  __shared__ __bf16 at[128 * AS];
  __shared__ __bf16 wt[64 * AS];
  const int tid  = threadIdx.x;
  const int lane = tid & 31;
  const int wave = tid >> 5;
  const int hi   = (lane >> 4) & 1;
  const int mBase = blockIdx.y * 128;
  const int nBase = blockIdx.x * 64;

  v8f acc[4] = {};

  for (int kb = 0; kb < D_; kb += 32) {
    __syncthreads();
#if USE_TDM
    if (wave == 0)                             // DMA A tile while others convert W
      tdm_load_2d_bf16(Xb + mBase * D_ + kb, at, 32, 128, D_, /*interval 16dw*/3, 0);
#else
#pragma unroll
    for (int i = 0; i < 4; ++i) {
      int idx = i * 256 + tid;
      int r = idx >> 3, c = (idx & 7) * 4;
      unsigned long long d = *(const unsigned long long*)&Xb[(mBase + r) * D_ + kb + c];
      *(unsigned*)&at[r * AS + c]     = (unsigned)d;
      *(unsigned*)&at[r * AS + c + 2] = (unsigned)(d >> 32);
    }
#endif
#pragma unroll
    for (int i = 0; i < 2; ++i) {
      int idx = i * 256 + tid;
      int r = idx >> 3, c = (idx & 7) * 4;
      float4 f = *(const float4*)&W[(nBase + r) * D_ + kb + c];
      *(unsigned*)&wt[r * AS + c]     = pack_bf16x2(f.x, f.y);
      *(unsigned*)&wt[r * AS + c + 2] = pack_bf16x2(f.z, f.w);
    }
    if (kb + 32 < D_ && tid >= 192)
      __builtin_prefetch(&W[(nBase + tid - 192) * D_ + kb + 32], 0, 3);
#if USE_TDM
    if (wave == 0) __builtin_amdgcn_s_wait_tensorcnt(0);
#endif
    __syncthreads();

    const int mrow = wave * 16 + (lane & 15);
    v16bf a = load_a_frag(at, mrow, AS, 0, hi);
    v16bf bf4[4];
#pragma unroll
    for (int t = 0; t < 4; ++t)
      bf4[t] = load_b_frag(wt, t * 16 + (lane & 15), AS, 0, hi);
#pragma unroll
    for (int t = 0; t < 4; ++t)
      acc[t] = wmma_bf16(a, bf4[t], acc[t]);
  }

  const int nloc = lane & 15;
#pragma unroll
  for (int t = 0; t < 4; ++t) {
    int n = nBase + t * 16 + nloc;
    float bv = bias[n];
#pragma unroll
    for (int r = 0; r < 8; ++r) {
      int m = mBase + wave * 16 + r + hi * 8;
      out[m * D_ + n] = acc[t][r] + bv;
    }
  }
}

extern "C" void kernel_launch(void* const* d_in, const int* in_sizes, int n_in,
                              void* d_out, int out_size, void* d_ws, size_t ws_size,
                              hipStream_t stream) {
  (void)in_sizes; (void)n_in; (void)out_size; (void)ws_size;
  // setup_inputs order: v, k, q, Wq, bq, Wk, bk, Wv, bv, Wo, bo (all f32)
  const float* v  = (const float*)d_in[0];
  const float* k  = (const float*)d_in[1];
  const float* q  = (const float*)d_in[2];
  const float* Wq = (const float*)d_in[3];
  const float* bq = (const float*)d_in[4];
  const float* Wk = (const float*)d_in[5];
  const float* bk = (const float*)d_in[6];
  const float* Wv = (const float*)d_in[7];
  const float* bv = (const float*)d_in[8];
  const float* Wo = (const float*)d_in[9];
  const float* bo = (const float*)d_in[10];
  float* out = (float*)d_out;

  // Workspace: 4x 8MB bf16 buffers (q2,k2,v2 in (B,H,S,DK); att in (B,S,D))
  char* ws = (char*)d_ws;
  const size_t BUF = (size_t)B_ * H_ * S_ * DK_ * sizeof(__bf16);
  __bf16* q2b  = (__bf16*)(ws + 0 * BUF);
  __bf16* k2b  = (__bf16*)(ws + 1 * BUF);
  __bf16* v2b  = (__bf16*)(ws + 2 * BUF);
  __bf16* attb = (__bf16*)(ws + 3 * BUF);

  dim3 gp(D_ / 64, (B_ * S_) / 128);  // 16 x 32
  proj_kernel<<<gp, 256, 0, stream>>>(q, Wq, bq, q2b);
  proj_kernel<<<gp, 256, 0, stream>>>(k, Wk, bk, k2b);
  proj_kernel<<<gp, 256, 0, stream>>>(v, Wv, bv, v2b);

  attn_kernel<<<dim3(S_ / 64, H_, B_), 128, 0, stream>>>(q2b, k2b, v2b, attb);

  oproj_kernel<<<gp, 256, 0, stream>>>(attb, Wo, bo, out);
}
